// GINModel_62362925138436
// MI455X (gfx1250) — matrix operand order repaired and checked by reading, not compiled, not run on hardware
//
#include <hip/hip_runtime.h>

#define DDIM 128

typedef __attribute__((ext_vector_type(16))) __bf16 v16bf;
typedef __attribute__((ext_vector_type(8)))  float  v8f;

// Fragment-packed weight region layout (in d_ws, after the two ping-pong bufs):
//   per matrix: [hi: 16384 __bf16][lo: 16384 __bf16]
//   matrix index mat = l*2 + g  (g=0 -> W1, g=1 -> W2)
//   fragment f = c*8 + t (k-chunk c in 0..3, n-tile t in 0..7)
//   element j of lane's v16bf at:  mat*32768 + part*16384 + (f*32 + lane)*16 + j
#define WFRAG_PER_MAT 32768  // hi + lo halves

// ---------------------------------------------------------------------------
// agg = x   (the "+x" self term of GINConv, eps = 0)
// ---------------------------------------------------------------------------
__global__ void copy_f4_kernel(float4* __restrict__ dst, const float4* __restrict__ src,
                               long n4) {
  long i = (long)blockIdx.x * blockDim.x + threadIdx.x;
  if (i < n4) dst[i] = src[i];
}

__global__ void zero_kernel(float* __restrict__ p, int n) {
  int i = blockIdx.x * blockDim.x + threadIdx.x;
  if (i < n) p[i] = 0.f;
}

// ---------------------------------------------------------------------------
// Pack all layer weights into bf16 hi/lo, laid out exactly as WMMA B-fragments.
// One thread per (mat, c, t, lane); 16 elements -> one 32B v16bf store each.
// B layout: k = 32c + (j<8?0:16) + (lane<16?0:8) + (j&7),  n = 16t + (lane&15)
// ---------------------------------------------------------------------------
__global__ void pack_w_kernel(const float* __restrict__ Ws1, const float* __restrict__ Ws2,
                              __bf16* __restrict__ wf, int L) {
  int tid = blockIdx.x * blockDim.x + threadIdx.x;
  int lane = tid & 31;
  int f    = (tid >> 5) & 31;   // c*8 + t
  int mat  = tid >> 10;         // l*2 + g
  if (mat >= 2 * L) return;
  int l = mat >> 1;
  int g = mat & 1;
  const float* W = (g == 0 ? Ws1 : Ws2) + (size_t)l * DDIM * DDIM;
  int c = f >> 3;
  int t = f & 7;
  int koff = 32 * c + ((lane < 16) ? 0 : 8);
  int n    = 16 * t + (lane & 15);
  v16bf h, lo;
#pragma unroll
  for (int j = 0; j < 16; ++j) {
    float w = W[(koff + (j < 8 ? j : j + 8)) * DDIM + n];
    __bf16 wh = (__bf16)w;
    h[j]  = wh;
    lo[j] = (__bf16)(w - (float)wh);
  }
  __bf16* base = wf + (size_t)mat * WFRAG_PER_MAT + ((size_t)f * 32 + lane) * 16;
  *(v16bf*)(base)         = h;
  *(v16bf*)(base + 16384) = lo;
}

// ---------------------------------------------------------------------------
// agg[dst[e]] += x[src[e]]  — one wave32 per edge, float4 gather per lane,
// 4x global_atomic_add_f32. This is the bandwidth bound of the whole model.
// ---------------------------------------------------------------------------
__global__ void scatter_kernel(const float* __restrict__ x, float* __restrict__ agg,
                               const int* __restrict__ srcI, const int* __restrict__ dstI,
                               long E) {
  long gt = (long)blockIdx.x * blockDim.x + threadIdx.x;
  long e  = gt >> 5;
  int lane = (int)(gt & 31);
  if (e >= E) return;
  long s = srcI[e];
  long d = dstI[e];
  const float4 v = *(const float4*)(x + s * DDIM + lane * 4);
  float* q = agg + d * DDIM + lane * 4;
  unsafeAtomicAdd(q + 0, v.x);
  unsafeAtomicAdd(q + 1, v.y);
  unsafeAtomicAdd(q + 2, v.z);
  unsafeAtomicAdd(q + 3, v.w);
}

// ---------------------------------------------------------------------------
// A-fragment loader (16x32 bf16 hi/lo split). Per k-chunk, each lane's 16
// elements are two contiguous 8-float runs -> 4x float4 (b128) loads.
//   k(j) = 32c + (j<8?0:16) + off8 + (j&7),  m = lane & 15
// ---------------------------------------------------------------------------
__device__ __forceinline__ void load_a_frags(const float* rowp, int off8,
                                             v16bf* ahi, v16bf* alo) {
#pragma unroll
  for (int c = 0; c < 4; ++c) {
    const int kb = 32 * c;
    float f[16];
    *(float4*)(f + 0)  = *(const float4*)(rowp + kb + off8);
    *(float4*)(f + 4)  = *(const float4*)(rowp + kb + off8 + 4);
    *(float4*)(f + 8)  = *(const float4*)(rowp + kb + 16 + off8);
    *(float4*)(f + 12) = *(const float4*)(rowp + kb + 16 + off8 + 4);
    v16bf h, l;
#pragma unroll
    for (int j = 0; j < 16; ++j) {
      __bf16 fh = (__bf16)f[j];
      h[j] = fh;
      l[j] = (__bf16)(f[j] - (float)fh);
    }
    ahi[c] = h;
    alo[c] = l;
  }
}

// One 16x16 output tile: fp32-emulated bf16 GEMM (AhBh + AhBl + AlBh).
// B fragments come pre-packed: one 32B v16bf load each for hi and lo.
__device__ __forceinline__ v8f gemm_tile(const v16bf* ahi, const v16bf* alo,
                                         const __bf16* __restrict__ Wf, float binit,
                                         int t, int lane) {
  v8f acc;
#pragma unroll
  for (int v = 0; v < 8; ++v) acc[v] = binit;
#pragma unroll
  for (int c = 0; c < 4; ++c) {
    const __bf16* p = Wf + ((size_t)(c * 8 + t) * 32 + lane) * 16;
    const v16bf bh = *(const v16bf*)(p);
    const v16bf bl = *(const v16bf*)(p + 16384);
    acc = __builtin_amdgcn_wmma_f32_16x16x32_bf16(false, ahi[c], false, bh,
                                                  (short)0, acc, false, false);
    acc = __builtin_amdgcn_wmma_f32_16x16x32_bf16(false, ahi[c], false, bl,
                                                  (short)0, acc, false, false);
    acc = __builtin_amdgcn_wmma_f32_16x16x32_bf16(false, alo[c], false, bh,
                                                  (short)0, acc, false, false);
  }
  return acc;
}

// ---------------------------------------------------------------------------
// Fused MLP:  data <- relu(data @ W1 + b1) @ W2 + b2   (in place, per 16-row tile)
// 4 waves / block, each wave owns one 16x128 row tile.
// ---------------------------------------------------------------------------
__global__ void __launch_bounds__(128)
mlp_kernel(float* data,
           const __bf16* __restrict__ w1f, const float* __restrict__ b1,
           const __bf16* __restrict__ w2f, const float* __restrict__ b2, int nrows) {
  __shared__ float hbuf[4][16 * DDIM];  // per-wave staging of h (32 KB)

  const int lane = threadIdx.x & 31;
  const int wave = threadIdx.x >> 5;
  const int tile_row = (blockIdx.x * 4 + wave) * 16;
  const bool tileok = (tile_row + 16) <= nrows;  // wave-uniform: EXEC stays full
  const int m    = lane & 15;
  const int off8 = (lane < 16) ? 0 : 8;

  v16bf ahi[4], alo[4];

  if (tileok) {
    // ---- GEMM1: h = relu((x+agg) @ W1 + b1) -> LDS ----
    const float* rowp = data + (size_t)(tile_row + m) * DDIM;
    load_a_frags(rowp, off8, ahi, alo);
#pragma unroll 1
    for (int t = 0; t < 8; ++t) {
      int nb = 16 * t;
      v8f acc = gemm_tile(ahi, alo, w1f, b1[nb + m], t, lane);
#pragma unroll
      for (int v = 0; v < 8; ++v) {
        float r = acc[v] > 0.f ? acc[v] : 0.f;
        // C/D layout: row = off8 + v, col = nb + m
        hbuf[wave][(off8 + v) * DDIM + nb + m] = r;
      }
    }
  }
  __syncthreads();
  if (tileok) {
    // ---- GEMM2: x' = h @ W2 + b2 -> global (in place over agg) ----
    const float* hrow = &hbuf[wave][m * DDIM];
    load_a_frags(hrow, off8, ahi, alo);
    float* orow = data + (size_t)tile_row * DDIM;
#pragma unroll 1
    for (int t = 0; t < 8; ++t) {
      int nb = 16 * t;
      v8f acc = gemm_tile(ahi, alo, w2f, b2[nb + m], t, lane);
#pragma unroll
      for (int v = 0; v < 8; ++v)
        orow[(size_t)(off8 + v) * DDIM + nb + m] = acc[v];
    }
  }
}

// ---------------------------------------------------------------------------
// global_add_pool: out[batch[i]] += x[i]  — one wave per node
// ---------------------------------------------------------------------------
__global__ void pool_kernel(const float* __restrict__ x, const int* __restrict__ batch,
                            float* __restrict__ out, int n) {
  long gt = (long)blockIdx.x * blockDim.x + threadIdx.x;
  long i  = gt >> 5;
  int lane = (int)(gt & 31);
  if (i >= n) return;
  long g = batch[i];
  const float4 v = *(const float4*)(x + i * DDIM + lane * 4);
  float* q = out + g * DDIM + lane * 4;
  unsafeAtomicAdd(q + 0, v.x);
  unsafeAtomicAdd(q + 1, v.y);
  unsafeAtomicAdd(q + 2, v.z);
  unsafeAtomicAdd(q + 3, v.w);
}

// ---------------------------------------------------------------------------
extern "C" void kernel_launch(void* const* d_in, const int* in_sizes, int n_in,
                              void* d_out, int out_size, void* d_ws, size_t ws_size,
                              hipStream_t stream) {
  const float* x_in = (const float*)d_in[0];
  const int*   eidx = (const int*)d_in[1];   // [2, E] flat: row0 = src, row1 = dst
  const int*   batch = (const int*)d_in[2];
  const float* Ws1 = (const float*)d_in[3];  // [L, D, D]
  const float* bs1 = (const float*)d_in[4];  // [L, D]
  const float* Ws2 = (const float*)d_in[5];
  const float* bs2 = (const float*)d_in[6];
  float* out = (float*)d_out;

  const int  N = in_sizes[0] / DDIM;           // 50000
  const long E = (long)in_sizes[1] / 2;        // 1.6M
  const int  L = in_sizes[3] / (DDIM * DDIM);  // 4
  const int  G = out_size / DDIM;              // 512

  const int* srcI = eidx;
  const int* dstI = eidx + E;

  const long nd = (long)N * DDIM;
  float* bufs[2] = { (float*)d_ws, (float*)d_ws + nd };   // ping-pong, 2 x 25.6 MB
  __bf16* wf = (__bf16*)((float*)d_ws + 2 * nd);          // +512 KB packed weights
  (void)n_in; (void)ws_size;

  // Pack all layer weights into bf16 hi/lo fragment layout (once per call).
  {
    int threads = 2 * L * 1024;  // (mat, c*8+t, lane)
    pack_w_kernel<<<(threads + 255) / 256, 256, 0, stream>>>(Ws1, Ws2, wf, L);
  }

  const float* xcur = x_in;
  for (int l = 0; l < L; ++l) {
    float* agg = bufs[l & 1];
    const long n4 = nd / 4;
    copy_f4_kernel<<<(unsigned)((n4 + 255) / 256), 256, 0, stream>>>(
        (float4*)agg, (const float4*)xcur, n4);
    const long st = E * 32;
    scatter_kernel<<<(unsigned)((st + 255) / 256), 256, 0, stream>>>(
        xcur, agg, srcI, dstI, E);
    const int tiles = N / 16;                 // 3125, exact
    const int blocks = (tiles + 3) / 4;       // 782 (trailing waves idle-but-sync)
    mlp_kernel<<<blocks, 128, 0, stream>>>(
        agg,
        wf + (size_t)(l * 2 + 0) * WFRAG_PER_MAT, bs1 + (size_t)l * DDIM,
        wf + (size_t)(l * 2 + 1) * WFRAG_PER_MAT, bs2 + (size_t)l * DDIM, N);
    xcur = agg;
  }

  zero_kernel<<<(G * DDIM + 255) / 256, 256, 0, stream>>>(out, G * DDIM);
  const long pt = (long)N * 32;
  pool_kernel<<<(unsigned)((pt + 255) / 256), 256, 0, stream>>>(xcur, batch, out, N);
}
